// MLP_52269751992812
// MI455X (gfx1250) — matrix operand order repaired and checked by reading, not compiled, not run on hardware
//
#include <hip/hip_runtime.h>

// ---------------------------------------------------------------------------
// Types for CDNA5 WMMA / TDM (wave32)
// ---------------------------------------------------------------------------
typedef __attribute__((ext_vector_type(16))) __bf16 bf16x16;
typedef __attribute__((ext_vector_type(8)))  __bf16 bf16x8;
typedef __attribute__((ext_vector_type(8)))  float  f32x8;
typedef __attribute__((ext_vector_type(4)))  unsigned int u32x4;
typedef __attribute__((ext_vector_type(8)))  int i32x8;
typedef __attribute__((ext_vector_type(4)))  int i32x4;

#if defined(__gfx1250__) && __has_builtin(__builtin_amdgcn_tensor_load_to_lds) && \
    __has_builtin(__builtin_amdgcn_s_wait_tensorcnt)
#define USE_TDM 1
#else
#define USE_TDM 0
#endif

static __device__ __forceinline__ __bf16 f2bf(float f) {
    // round-to-nearest-even f32 -> bf16
    unsigned u = __builtin_bit_cast(unsigned, f);
    unsigned r = u + 0x7FFFu + ((u >> 16) & 1u);
    unsigned short h = (unsigned short)(r >> 16);
    return __builtin_bit_cast(__bf16, h);
}

#define BM 128
#define BN 128
#define BKK 32
#define PK 40   // LDS K-stride in bf16 elems: 32 data + 8 pad (80B, 16B-aligned)

#if USE_TDM
// Issue one TDM 2D tile load: 128 rows x 32 bf16 (64B) per row, row stride
// `ld` elems in global, packed into LDS rows of PK=40 bf16 via the TDM pad
// feature (pad_interval=3 -> every 16 DWORDs, pad_amount=3 -> 4 DWORDs = 16B).
// D# built per CDNA5 ISA §8.3/8.4 (group0: count=1, lds_addr, global_addr,
// type=2; group1: data_size=2B, pad_enable, dims/strides). 2D -> groups 2/3
// (and the trailing group of the 6-arg builtin) are zero.
static __device__ __forceinline__ void tdm_issue(const __bf16* gp, unsigned ldsOff, unsigned ld)
{
    unsigned long long ga = (unsigned long long)(size_t)gp;
    u32x4 g0 = { 1u,                                   // count=1 (user descriptor)
                 ldsOff,                               // lds_addr (bytes)
                 (unsigned)ga,                         // global_addr[31:0]
                 (unsigned)(ga >> 32) | (2u << 30) };  // global_addr[56:32] | type=2
    i32x8 g1 = { (int)((1u << 16) | (1u << 20) | (3u << 22) | (3u << 25)),
                 //    data_size=2B | pad_enable | pad_interval=16dw | pad_amount=4dw
                 (int)((ld & 0xFFFFu) << 16),          // tensor_dim0 lo16 (bits 63:48)
                 (int)(ld >> 16),                      // tensor_dim0 hi16, tensor_dim1 lo16=0
                 (int)(0x10u | (32u << 16)),           // tensor_dim1 hi16 (=1M rows), tile_dim0=32
                 (int)128,                             // tile_dim1=128, tile_dim2=0
                 (int)ld,                              // tensor_dim0_stride lo32 (elems)
                 0, 0 };
    i32x4 z4 = {0, 0, 0, 0};
    i32x8 z8 = {0, 0, 0, 0, 0, 0, 0, 0};
    __builtin_amdgcn_tensor_load_to_lds(g0, g1, z4, z4, z8, 0);
}
#endif

// ---------------------------------------------------------------------------
// Tiled GEMM: C[M,N] = epi(A[M,K] @ Bt^T), Bt pre-transposed bf16 [N][Kpad].
// A is either f32 (converted while staging, K bounds vs Kreal) or pre-padded
// bf16 [M][lda] (TDM-loaded straight to LDS when available).
// Epilogue: +bias[n], ReLU, * mul[m,n], optional f32 and/or bf16 stores.
// Block 256 threads (8 waves), tile 128x128x32, double-buffered LDS,
// v_wmma_f32_16x16x32_bf16 with f32 accumulation.
// ---------------------------------------------------------------------------
__global__ __launch_bounds__(256)
void gemm_bf16_wmma(const float* __restrict__ Af32, const __bf16* __restrict__ Abf, int lda,
                    const __bf16* __restrict__ Bt, int ldb,
                    const float* __restrict__ bias, const float* __restrict__ mul,
                    float* __restrict__ C, __bf16* __restrict__ Cb,
                    int Ndim, int Kreal, int Kpad, int do_relu)
{
    __shared__ __bf16 As[2][BM][PK];
    __shared__ __bf16 Bs[2][BN][PK];

    const int tid  = threadIdx.x;
    const int lane = tid & 31;
    const int wave = tid >> 5;          // 0..7
    const int wm   = wave >> 1;         // 0..3 : 32-row group
    const int wn   = wave & 1;          // 0..1 : 64-col group
    const int row0 = blockIdx.y * BM;
    const int col0 = blockIdx.x * BN;
    const int l15  = lane & 15;
    const int half = lane >> 4;
    const int nIter = Kpad / BKK;

    f32x8 acc[2][4];
    const f32x8 zero = {0.f,0.f,0.f,0.f,0.f,0.f,0.f,0.f};
#pragma unroll
    for (int i = 0; i < 2; ++i)
#pragma unroll
        for (int j = 0; j < 4; ++j) acc[i][j] = zero;

#if USE_TDM
    const unsigned asOff0 = (unsigned)(size_t)&As[0][0][0];
    const unsigned asOff1 = (unsigned)(size_t)&As[1][0][0];
    const unsigned bsOff0 = (unsigned)(size_t)&Bs[0][0][0];
    const unsigned bsOff1 = (unsigned)(size_t)&Bs[1][0][0];
    if (wave == 0) {   // prologue: DMA tile 0 into buffer 0 (TDM ignores EXEC; 1 issue/wave)
        if (Abf) tdm_issue(Abf + (size_t)row0 * lda, asOff0, (unsigned)lda);
        tdm_issue(Bt + (size_t)col0 * ldb, bsOff0, (unsigned)ldb);
    }
#endif

    for (int it = 0; it < nIter; ++it) {
        const int p  = it & 1;
        const int k0 = it * BKK;
#if USE_TDM
        if (wave == 0) {
            if (it + 1 < nIter) {       // async prefetch of next tile into other buffer
                const int kn = k0 + BKK;
                const unsigned ao = p ? asOff0 : asOff1;
                const unsigned bo = p ? bsOff0 : bsOff1;
                if (Abf) tdm_issue(Abf + (size_t)row0 * lda + kn, ao, (unsigned)lda);
                tdm_issue(Bt + (size_t)col0 * ldb + kn, bo, (unsigned)ldb);
                // retire only the current tile's DMAs (in-order per wave)
                if (Abf) __builtin_amdgcn_s_wait_tensorcnt(2);
                else     __builtin_amdgcn_s_wait_tensorcnt(1);
            } else {
                __builtin_amdgcn_s_wait_tensorcnt(0);
            }
        }
        if (!Abf) {                     // f32 A: convert while staging (16 elems/thread)
#pragma unroll
            for (int t = 0; t < 16; ++t) {
                int e = t * 256 + tid;
                int m = e >> 5, k = e & 31;
                int gk = k0 + k;
                float v = (gk < Kreal) ? Af32[(size_t)(row0 + m) * lda + gk] : 0.0f;
                As[p][m][k] = f2bf(v);
            }
            if (k0 + BKK < Kpad)
                __builtin_prefetch(&Af32[(size_t)(row0 + (tid >> 1)) * lda + k0 + BKK], 0, 1);
        }
#else
        // ---- manual fallback staging ----
        if (Abf) {
#pragma unroll
            for (int t = 0; t < 16; ++t) {
                int e = t * 256 + tid;
                int m = e >> 5, k = e & 31;
                As[p][m][k] = Abf[(size_t)(row0 + m) * lda + k0 + k];
            }
        } else {
#pragma unroll
            for (int t = 0; t < 16; ++t) {
                int e = t * 256 + tid;
                int m = e >> 5, k = e & 31;
                int gk = k0 + k;
                float v = (gk < Kreal) ? Af32[(size_t)(row0 + m) * lda + gk] : 0.0f;
                As[p][m][k] = f2bf(v);
            }
        }
#pragma unroll
        for (int t = 0; t < 16; ++t) {
            int e = t * 256 + tid;
            int n = e >> 5, k = e & 31;
            Bs[p][n][k] = Bt[(size_t)(col0 + n) * ldb + k0 + k];
        }
#endif
        __syncthreads();

        // A fragments per ISA 16-bit 16x32 layout:
        // lanes 0-15: e0..7 -> K 0..7,  e8..15 -> K 16..23
        // lanes16-31: e0..7 -> K 8..15, e8..15 -> K 24..31
        bf16x16 afrag[2];
#pragma unroll
        for (int i = 0; i < 2; ++i) {
            int mrow = wm * 32 + i * 16 + l15;
            bf16x8 lo = *(const bf16x8*)&As[p][mrow][half * 8];
            bf16x8 hi = *(const bf16x8*)&As[p][mrow][16 + half * 8];
            afrag[i] = __builtin_shufflevector(lo, hi, 0,1,2,3,4,5,6,7,8,9,10,11,12,13,14,15);
        }
        // B fragments: same K layout as A (lanes 0-15: K 0..7,16..23; lanes 16-31: K 8..15,24..31)
#pragma unroll
        for (int j = 0; j < 4; ++j) {
            int ncol = wn * 64 + j * 16 + l15;
            bf16x8 lo = *(const bf16x8*)&Bs[p][ncol][half * 8];
            bf16x8 hi = *(const bf16x8*)&Bs[p][ncol][16 + half * 8];
            bf16x16 bfrag = __builtin_shufflevector(lo, hi, 0,1,2,3,4,5,6,7,8,9,10,11,12,13,14,15);
#pragma unroll
            for (int i = 0; i < 2; ++i)
                acc[i][j] = __builtin_amdgcn_wmma_f32_16x16x32_bf16(
                    false, afrag[i], false, bfrag, (short)0, acc[i][j], false, false);
        }
        __syncthreads();
    }

    // Epilogue. C/D layout: lane n = l15; VGPR v -> M = v + 8*half
#pragma unroll
    for (int i = 0; i < 2; ++i) {
        int mbase = row0 + wm * 32 + i * 16 + half * 8;
#pragma unroll
        for (int j = 0; j < 4; ++j) {
            int n = col0 + wn * 64 + j * 16 + l15;
            float bv = bias ? bias[n] : 0.0f;
#pragma unroll
            for (int v = 0; v < 8; ++v) {
                int m = mbase + v;
                float o = acc[i][j][v] + bv;
                if (do_relu) o = o > 0.f ? o : 0.f;
                if (mul) o *= mul[(size_t)m * Ndim + n];
                if (C)  C[(size_t)m * Ndim + n] = o;
                if (Cb) Cb[(size_t)m * Ndim + n] = f2bf(o);
            }
        }
    }
}

// out[n*Kpad + k] = (k < K) ? bf16(W[k*N + n]) : 0   (transpose + pad + convert)
__global__ __launch_bounds__(256)
void transpose_to_bf16(const float* __restrict__ W, __bf16* __restrict__ out,
                       int K, int N, int Kpad)
{
    int idx = blockIdx.x * 256 + threadIdx.x;
    if (idx >= N * Kpad) return;
    int n = idx / Kpad, k = idx - n * Kpad;
    float v = (k < K) ? W[(size_t)k * N + n] : 0.0f;
    out[(size_t)n * Kpad + k] = f2bf(v);
}

__global__ __launch_bounds__(256)
void copy_to_bf16(const float* __restrict__ src, __bf16* __restrict__ dst, int n)
{
    int i = blockIdx.x * 256 + threadIdx.x;
    if (i < n) dst[i] = f2bf(src[i]);
}

// ---------------------------------------------------------------------------
// Row softmax (axis=1) of +score / -score + row argmax. M = 1024 cols fixed.
// ---------------------------------------------------------------------------
__global__ __launch_bounds__(256)
void row_softmax_argmax(const float* __restrict__ score,
                        float* __restrict__ sm_out, float* __restrict__ usm_out,
                        int* __restrict__ g_out)
{
    const int M = 1024;
    const int row = blockIdx.x;
    const int tid = threadIdx.x;
    const float* s = score + (size_t)row * M;

    float v[4];
    float mx = -3.4e38f, mn = 3.4e38f, bv = -3.4e38f;
    int bi = M;
#pragma unroll
    for (int t = 0; t < 4; ++t) {
        int c = tid + t * 256;
        float x = s[c];
        v[t] = x;
        mx = fmaxf(mx, x);
        mn = fminf(mn, x);
        if (x > bv || (x == bv && c < bi)) { bv = x; bi = c; }
    }
    __shared__ float sA[256], sB[256], sV[256];
    __shared__ int   sI[256];
    sA[tid] = mx; sB[tid] = mn; sV[tid] = bv; sI[tid] = bi;
    __syncthreads();
    for (int o = 128; o > 0; o >>= 1) {
        if (tid < o) {
            sA[tid] = fmaxf(sA[tid], sA[tid + o]);
            sB[tid] = fminf(sB[tid], sB[tid + o]);
            float ov = sV[tid + o]; int oi = sI[tid + o];
            if (ov > sV[tid] || (ov == sV[tid] && oi < sI[tid])) { sV[tid] = ov; sI[tid] = oi; }
        }
        __syncthreads();
    }
    float rmx = sA[0], rmn = sB[0];
    if (tid == 0) g_out[row] = sI[0];
    __syncthreads();

    float sp = 0.f, sn = 0.f;
#pragma unroll
    for (int t = 0; t < 4; ++t) { sp += __expf(v[t] - rmx); sn += __expf(rmn - v[t]); }
    sA[tid] = sp; sB[tid] = sn;
    __syncthreads();
    for (int o = 128; o > 0; o >>= 1) {
        if (tid < o) { sA[tid] += sA[tid + o]; sB[tid] += sB[tid + o]; }
        __syncthreads();
    }
    float rp = 1.f / sA[0], rn = 1.f / sB[0];
    float* smr  = sm_out  + (size_t)row * M;
    float* usmr = usm_out + (size_t)row * M;
#pragma unroll
    for (int t = 0; t < 4; ++t) {
        int c = tid + t * 256;
        smr[c]  = __expf(v[t] - rmx) * rp;
        usmr[c] = __expf(rmn - v[t]) * rn;
    }
}

// ---------------------------------------------------------------------------
// Column softmax (axis=0) of +score / -score; one thread per column (coalesced).
// ---------------------------------------------------------------------------
__global__ __launch_bounds__(256)
void col_softmax(const float* __restrict__ score,
                 float* __restrict__ sq_out, float* __restrict__ usq_out,
                 float* __restrict__ colmax_out, int Nrows, int Mcols)
{
    const int col = blockIdx.x * 256 + threadIdx.x;
    float mx = -3.4e38f, mn = 3.4e38f;
    for (int i = 0; i < Nrows; ++i) {
        float v = score[(size_t)i * Mcols + col];
        mx = fmaxf(mx, v); mn = fminf(mn, v);
    }
    float sp = 0.f, sn = 0.f;
    for (int i = 0; i < Nrows; ++i) {
        float v = score[(size_t)i * Mcols + col];
        sp += __expf(v - mx); sn += __expf(mn - v);
    }
    float rp = 1.f / sp, rn = 1.f / sn;
    for (int i = 0; i < Nrows; ++i) {
        float v = score[(size_t)i * Mcols + col];
        sq_out [(size_t)i * Mcols + col] = __expf(v - mx) * rp;
        usq_out[(size_t)i * Mcols + col] = __expf(mn - v) * rn;
    }
    colmax_out[col] = mx;
}

// w[i] = sq[i,g]/max_i' sq[i',g] = exp(score[i,g] - colmax[g])
__global__ __launch_bounds__(256)
void compute_w(const float* __restrict__ score, const int* __restrict__ g,
               const float* __restrict__ colmax, float* __restrict__ w, int Mcols)
{
    int i = blockIdx.x * 256 + threadIdx.x;
    int gi = g[i];
    w[i] = __expf(score[(size_t)i * Mcols + gi] - colmax[gi]);
}

// ---------------------------------------------------------------------------
// Deterministic segment-sum (block per slot) fused with L2 row-normalize.
// ---------------------------------------------------------------------------
__global__ __launch_bounds__(256)
void segment_normalize(const float* __restrict__ h, const float* __restrict__ w,
                       const int* __restrict__ g, const float* __restrict__ Kmem,
                       float* __restrict__ newmem, int Nrows)
{
    const int H = 512;
    const int slot = blockIdx.x;
    const int tid  = threadIdx.x;
    float acc0 = 0.f, acc1 = 0.f;
    for (int i = 0; i < Nrows; ++i) {
        if (g[i] == slot) {                      // uniform -> scalar branch
            float wi = w[i];
            const float* hr = h + (size_t)i * H;
            acc0 += wi * hr[tid];
            acc1 += wi * hr[tid + 256];
        }
    }
    float v0 = acc0 + Kmem[(size_t)slot * H + tid];
    float v1 = acc1 + Kmem[(size_t)slot * H + tid + 256];
    __shared__ float red[256];
    red[tid] = v0 * v0 + v1 * v1;
    __syncthreads();
    for (int o = 128; o > 0; o >>= 1) { if (tid < o) red[tid] += red[tid + o]; __syncthreads(); }
    float inv = 1.f / fmaxf(sqrtf(red[0]), 1e-12f);
    newmem[(size_t)slot * H + tid]       = v0 * inv;
    newmem[(size_t)slot * H + tid + 256] = v1 * inv;
}

// pred[i] = h[i,:] . Wd + bd   (one wave32 per row)
__global__ __launch_bounds__(256)
void pred_gemv(const float* __restrict__ h, const float* __restrict__ Wd,
               const float* __restrict__ bd, float* __restrict__ pred)
{
    const int H = 512;
    int row  = blockIdx.x * 8 + (threadIdx.x >> 5);
    int lane = threadIdx.x & 31;
    const float* hr = h + (size_t)row * H;
    float s = 0.f;
#pragma unroll
    for (int t = 0; t < 16; ++t) s += hr[lane + t * 32] * Wd[lane + t * 32];
    for (int o = 16; o > 0; o >>= 1) s += __shfl_down(s, o, 32);
    if (lane == 0) pred[row] = s + bd[0];
}

// ---------------------------------------------------------------------------
extern "C" void kernel_launch(void* const* d_in, const int* in_sizes, int n_in,
                              void* d_out, int out_size, void* d_ws, size_t ws_size,
                              hipStream_t stream) {
    (void)in_sizes; (void)n_in; (void)out_size; (void)ws_size;
    const float* x  = (const float*)d_in[0];
    const float* Km = (const float*)d_in[1];
    const float* W0 = (const float*)d_in[2];
    const float* b0 = (const float*)d_in[3];
    const float* W1 = (const float*)d_in[4];
    const float* b1 = (const float*)d_in[5];
    const float* W2 = (const float*)d_in[6];
    const float* b2 = (const float*)d_in[7];
    const float* Wd = (const float*)d_in[8];
    const float* bd = (const float*)d_in[9];

    const int N = 32768, D = 360, Dp = 384, H = 512, M = 1024;

    // outputs: (pred, upd_n, usq, usm, upd_p, sq, sm, new_mem) concatenated
    float* out    = (float*)d_out;
    float* pred   = out;
    float* updn   = pred + N;
    float* usq    = updn + (size_t)N * H;
    float* usm    = usq  + (size_t)N * M;
    float* updp   = usm  + (size_t)N * M;
    float* sq     = updp + (size_t)N * H;
    float* sm     = sq   + (size_t)N * M;
    float* newmem = sm   + (size_t)N * M;

    // workspace
    float*  ws     = (float*)d_ws;
    float*  hbuf   = ws;                          // N*H f32 (final h)
    float*  score  = hbuf + (size_t)N * H;        // N*M f32
    float*  colmax = score + (size_t)N * M;       // M
    int*    g      = (int*)(colmax + M);          // N
    float*  wrow   = (float*)(g + N);             // N
    __bf16* hbA    = (__bf16*)(wrow + N);         // N*H bf16 (h0, later h)
    __bf16* hbB    = hbA + (size_t)N * H;         // N*H bf16 (h1)
    __bf16* w0t    = hbB + (size_t)N * H;         // H*Dp  [n][k], padded
    __bf16* w1t    = w0t + (size_t)H * Dp;        // H*H
    __bf16* w2t    = w1t + (size_t)H * H;         // H*H
    __bf16* kmb    = w2t + (size_t)H * H;         // M*H  (Km as-is: B for score)
    __bf16* kmtT   = kmb + (size_t)M * H;         // H*M  (Km transposed: B for sm@K)

    dim3 blk(256);
    // --- one-time (per call) operand conversion/transposition to bf16 ---
    transpose_to_bf16<<<dim3((H * Dp) / 256), blk, 0, stream>>>(W0, w0t, D, H, Dp);
    transpose_to_bf16<<<dim3((H * H) / 256), blk, 0, stream>>>(W1, w1t, H, H, H);
    transpose_to_bf16<<<dim3((H * H) / 256), blk, 0, stream>>>(W2, w2t, H, H, H);
    copy_to_bf16<<<dim3((M * H) / 256), blk, 0, stream>>>(Km, kmb, M * H);
    transpose_to_bf16<<<dim3((H * M) / 256), blk, 0, stream>>>(Km, kmtT, M, H, M);

    // 1) h0 = relu(x @ W0 + b0)           (f32 A path, bf16 out)
    gemm_bf16_wmma<<<dim3(H / BN, N / BM), blk, 0, stream>>>(
        x, nullptr, D, w0t, Dp, b0, nullptr, nullptr, hbA, H, D, Dp, 1);
    // 2) h1 = relu(h0 @ W1 + b1)          (bf16 A via TDM)
    gemm_bf16_wmma<<<dim3(H / BN, N / BM), blk, 0, stream>>>(
        nullptr, hbA, H, w1t, H, b1, nullptr, nullptr, hbB, H, H, H, 1);
    // 3) h = relu(h1 @ W2 + b2)           (f32 + bf16 out)
    gemm_bf16_wmma<<<dim3(H / BN, N / BM), blk, 0, stream>>>(
        nullptr, hbB, H, w2t, H, b2, nullptr, hbuf, hbA, H, H, H, 1);
    // 4) score = h @ K^T                  (kmb rows are already [slot][k])
    gemm_bf16_wmma<<<dim3(M / BN, N / BM), blk, 0, stream>>>(
        nullptr, hbA, H, kmb, H, nullptr, nullptr, score, nullptr, M, H, H, 0);
    // 5) sm/usm (axis=1) + argmax g
    row_softmax_argmax<<<dim3(N), blk, 0, stream>>>(score, sm, usm, g);
    // 6) sq/usq (axis=0) + column max
    col_softmax<<<dim3(M / 256), blk, 0, stream>>>(score, sq, usq, colmax, N, M);
    // 7) w[i]
    compute_w<<<dim3(N / 256), blk, 0, stream>>>(score, g, colmax, wrow, M);
    // 8) new_mem (deterministic segment-sum + normalize)
    segment_normalize<<<dim3(M), blk, 0, stream>>>(hbuf, wrow, g, Km, newmem, N);
    // 9) upd_p = h * (sm @ K)             (f32 A path; h-multiply fused)
    gemm_bf16_wmma<<<dim3(H / BN, N / BM), blk, 0, stream>>>(
        sm, nullptr, M, kmtT, M, nullptr, hbuf, updp, nullptr, H, M, M, 0);
    // 10) upd_n = h * (usm @ K)
    gemm_bf16_wmma<<<dim3(H / BN, N / BM), blk, 0, stream>>>(
        usm, nullptr, M, kmtT, M, nullptr, hbuf, updn, nullptr, H, M, M, 0);
    // 11) pred = h @ Wd + bd
    pred_gemv<<<dim3(N / 8), blk, 0, stream>>>(hbuf, Wd, bd, pred);
}
